// MambaLayer_6562710028933
// MI455X (gfx1250) — compile-verified, matrix operand order checked by reading, not run on hardware
//
#include <hip/hip_runtime.h>
#include <hip/hip_bf16.h>

// ---------------- problem constants ----------------
#define BATCH   8
#define DMODEL  192
#define DOUT    192
#define DSTATE  16
#define DCONV   4
#define DINNER  384       // 2*192
#define DTRANK  12
#define LSEQ    4096      // 64*64
#define BL      (BATCH*LSEQ)   // 32768 token rows
#define NPADXP  48        // dt_rank + 2*d_state = 44, padded to 48 for 16-wide tiles
#define NCHUNK  32
#define CLEN    128       // 32*128 = 4096
#define NSEQ    (BATCH*DINNER*DSTATE)  // 49152 scan sequences
#define MWAVES  8         // M tiles (waves) per workgroup in GEMM

typedef _Float16 v16h __attribute__((ext_vector_type(16)));
typedef _Float16 v8h  __attribute__((ext_vector_type(8)));
typedef float    v8f  __attribute__((ext_vector_type(8)));
typedef unsigned int v4u __attribute__((ext_vector_type(4)));
typedef int      v8i  __attribute__((ext_vector_type(8)));
typedef int      v4i  __attribute__((ext_vector_type(4)));

// ---------------- fragment loader (16-bit A/B layout, ISA 7.12.2) ----------
// lane&15 = matrix row (M for A, N for B); lane>>4 selects K-half.
// elements 0..7  = K: kk + 8*half + 0..7
// elements 8..15 = K: kk + 16 + 8*half + 0..7   (two contiguous 16B loads)
__device__ __forceinline__ v16h load_frag16(const _Float16* __restrict__ rowk, int half) {
    const v8h* p0 = reinterpret_cast<const v8h*>(rowk + 8 * half);
    const v8h* p1 = reinterpret_cast<const v8h*>(rowk + 16 + 8 * half);
    v8h lo = *p0, hi = *p1;
    v16h r;
#pragma unroll
    for (int i = 0; i < 8; ++i) { r[i] = lo[i]; r[8 + i] = hi[i]; }
    return r;
}

// ---------------- WMMA GEMM: Y[M,N] = X[M,K] * W[N,K]^T ----------------
// 256-thread block = 8 waves = 8 M-tiles sharing one 16*NT-wide N strip.
// Weight strip (16*NT rows x K) staged into LDS once per block by the
// Tensor Data Mover; B fragments then come from LDS (ds_load_b128).
// OUTMODE 0: f16 row-major   1: f32 row-major
// OUTMODE 2: f32 transposed into [B, N, LSEQ] (+bias) -- final projection
template<int NT, int OUTMODE>
__global__ void __launch_bounds__(256)
gemm_wmma(const _Float16* __restrict__ A, const _Float16* __restrict__ W,
          void* __restrict__ Out, const float* __restrict__ bias,
          int M, int N, int K, int ldOut)
{
    extern __shared__ _Float16 smem[];   // 16*NT rows x K halfs
    const int tid  = threadIdx.x;
    const int lane = tid & 31;
    const int wave = tid >> 5;
    const int half = (lane >> 4) & 1;
    const int ln16 = lane & 15;
    const int m0 = (blockIdx.x * MWAVES + wave) * 16;
    const int n0 = blockIdx.y * 16 * NT;
    const int NROWS = 16 * NT;

    // ---- TDM: stage W[n0 .. n0+NROWS) x K into LDS (wave 0 issues) ----
    if (wave == 0) {
        const unsigned la = (unsigned)(uintptr_t)smem;                 // LDS byte addr
        const unsigned long long ga =
            (unsigned long long)(uintptr_t)(W + (size_t)n0 * K);       // tile start
        // D# group0: count=1 | lds_addr | global_addr[56:0] | type=2
        v4u g0 = { 1u, la, (unsigned)(ga & 0xFFFFFFFFu),
                   (unsigned)((ga >> 32) & 0x1FFFFFFu) | (2u << 30) };
        // D# group1: data_size=1 (2B); tensor_dim0=K; tensor_dim1=NROWS;
        //            tile_dim0=K; tile_dim1=NROWS; tensor_dim0_stride=K
        v8i g1 = { (int)(1u << 16),
                   (int)(((unsigned)K & 0xFFFFu) << 16),
                   (int)((((unsigned)K >> 16) & 0xFFFFu) |
                         (((unsigned)NROWS & 0xFFFFu) << 16)),
                   (int)(((unsigned)K & 0xFFFFu) << 16),
                   (int)((unsigned)NROWS & 0xFFFFu),
                   (int)K, 0, 0 };
        v4i g2 = { 0, 0, 0, 0 };
        v4i g3 = { 0, 0, 0, 0 };
        v8i g4 = { 0, 0, 0, 0, 0, 0, 0, 0 };
        __builtin_amdgcn_tensor_load_to_lds(g0, g1, g2, g3, g4, 0);
        __builtin_amdgcn_s_wait_tensorcnt(0);
    }
    __syncthreads();

    v8f acc[NT];
    v8f zero = {};
#pragma unroll
    for (int t = 0; t < NT; ++t) acc[t] = zero;

    const _Float16* arow = A + (size_t)(m0 + ln16) * K;
    v16h af = load_frag16(arow, half);                  // pipelined A fragment
    for (int kk = 0; kk < K; kk += 32) {
        v16h bf[NT];
#pragma unroll
        for (int t = 0; t < NT; ++t)                    // all B frags before WMMAs
            bf[t] = load_frag16(smem + (size_t)(t * 16 + ln16) * K + kk, half);
        v16h afn = af;
        if (kk + 32 < K) afn = load_frag16(arow + kk + 32, half);
#pragma unroll
        for (int t = 0; t < NT; ++t)
            acc[t] = __builtin_amdgcn_wmma_f32_16x16x32_f16(
                false, af, false, bf[t], (short)0, acc[t], false, false);
        af = afn;
    }

#pragma unroll
    for (int t = 0; t < NT; ++t) {
        const int col = n0 + t * 16 + ln16;
        if (OUTMODE == 0) {
            _Float16* o = (_Float16*)Out;
#pragma unroll
            for (int r = 0; r < 8; ++r)
                o[(size_t)(m0 + 8 * half + r) * ldOut + col] = (_Float16)acc[t][r];
        } else if (OUTMODE == 1) {
            float* o = (float*)Out;
#pragma unroll
            for (int r = 0; r < 8; ++r)
                o[(size_t)(m0 + 8 * half + r) * ldOut + col] = acc[t][r];
        } else {
            // global row = b*LSEQ + l ; out[b][col][l]. Lane holds 8 consecutive l.
            float* o = (float*)Out;
            const int row0 = m0 + 8 * half;
            const int b = row0 >> 12;
            const int l = row0 & (LSEQ - 1);
            const float bv = bias ? bias[col] : 0.f;
            float* op = o + ((size_t)b * N + col) * LSEQ + l;
#pragma unroll
            for (int r = 0; r < 8; ++r) op[r] = acc[t][r] + bv;
        }
    }
}

// ---------------- weight conversion ----------------
__global__ void cvt_f16(const float* __restrict__ in, _Float16* __restrict__ out, int n) {
    int i = blockIdx.x * 256 + threadIdx.x;
    if (i < n) out[i] = (_Float16)in[i];
}
__global__ void cvt_f16_pad(const float* __restrict__ in, _Float16* __restrict__ out,
                            int n_in, int n_out) {
    int i = blockIdx.x * 256 + threadIdx.x;
    if (i < n_out) out[i] = (i < n_in) ? (_Float16)in[i] : (_Float16)0.f;
}

// ---------------- LayerNorm (+optional residual) -> f16, transposing [b,c,l]->[row,c]
__global__ void __launch_bounds__(256)
ln_kernel(const float* __restrict__ x, const float* __restrict__ ym,
          const float* __restrict__ skip, const float* __restrict__ w,
          const float* __restrict__ bvec, _Float16* __restrict__ out)
{
    const int row = blockIdx.x;
    const int b = row >> 12, l = row & (LSEQ - 1);
    const int c = threadIdx.x;
    float v = 0.f;
    if (c < DMODEL) {
        float xf = x[((size_t)b * DMODEL + c) * LSEQ + l];
        v = ym ? (ym[(size_t)row * DMODEL + c] + skip[0] * xf) : xf;
    }
    __shared__ float s1[256], s2[256];
    s1[c] = v; s2[c] = v * v;
    __syncthreads();
    for (int off = 128; off > 0; off >>= 1) {
        if (c < off) { s1[c] += s1[c + off]; s2[c] += s2[c + off]; }
        __syncthreads();
    }
    const float mu = s1[0] * (1.f / DMODEL);
    const float var = s2[0] * (1.f / DMODEL) - mu * mu;
    const float rs = rsqrtf(var + 1e-5f);
    if (c < DMODEL)
        out[(size_t)row * DMODEL + c] = (_Float16)((v - mu) * rs * w[c] + bvec[c]);
}

// ---------------- causal depthwise conv (k=4) + SiLU ----------------
__global__ void conv_silu(const _Float16* __restrict__ xz, const float* __restrict__ cw,
                          const float* __restrict__ cb, _Float16* __restrict__ xc)
{
    int idx = blockIdx.x * 256 + threadIdx.x;
    if (idx >= BL * DINNER) return;
    const int d = idx % DINNER;
    const int row = idx / DINNER;
    const int l = row & (LSEQ - 1);
    float acc = cb[d];
    const float4 wv = *(const float4*)(cw + d * 4);
    const float wk[4] = { wv.x, wv.y, wv.z, wv.w };
#pragma unroll
    for (int k = 0; k < DCONV; ++k) {
        int dl = l + k - (DCONV - 1);
        if (dl >= 0)
            acc += (float)xz[(size_t)(row + k - (DCONV - 1)) * (2 * DINNER) + d] * wk[k];
    }
    xc[idx] = (_Float16)(acc / (1.f + __expf(-acc)));
}

// ---------------- dt = softplus(dbc[:, :12] @ dt_proj_w^T + b) ----------------
__global__ void dt_softplus(const float* __restrict__ dbc, const float* __restrict__ dtw,
                            const float* __restrict__ dtb, float* __restrict__ dt)
{
    int idx = blockIdx.x * 256 + threadIdx.x;
    if (idx >= BL * DINNER) return;
    const int d = idx % DINNER;
    const int row = idx / DINNER;
    const float* dr = dbc + (size_t)row * NPADXP;
    const float* wr = dtw + d * DTRANK;
    float acc = dtb[d];
#pragma unroll
    for (int r = 0; r < DTRANK; ++r) acc += dr[r] * wr[r];
    dt[idx] = (acc > 20.f) ? acc : log1pf(__expf(acc));
}

// ---------------- chunked selective scan ----------------
__device__ __forceinline__ void load16(const float* __restrict__ p, float* v) {
    const float4* q = (const float4*)p;
    float4 a = q[0], b = q[1], c = q[2], d = q[3];
    v[0]=a.x; v[1]=a.y; v[2]=a.z; v[3]=a.w;
    v[4]=b.x; v[5]=b.y; v[6]=b.z; v[7]=b.w;
    v[8]=c.x; v[9]=c.y; v[10]=c.z; v[11]=c.w;
    v[12]=d.x; v[13]=d.y; v[14]=d.z; v[15]=d.w;
}

// Pass A: per-chunk affine summary (prod a, folded b) per (b,d,n)
__global__ void __launch_bounds__(32)
scan_passA(const float* __restrict__ dt, const _Float16* __restrict__ xc,
           const float* __restrict__ dbc, const float* __restrict__ A_log,
           float* __restrict__ Ap, float* __restrict__ Bc)
{
    const int lane = threadIdx.x;
    const int chunk = blockIdx.x & (NCHUNK - 1);
    const int dblk = (blockIdx.x >> 5) % (DINNER / 32);
    const int b = blockIdx.x / (NCHUNK * (DINNER / 32));
    const int d = dblk * 32 + lane;

    float An[DSTATE];
#pragma unroll
    for (int n = 0; n < DSTATE; ++n) An[n] = -__expf(A_log[d * DSTATE + n]);
    float ap[DSTATE], bc[DSTATE];
#pragma unroll
    for (int n = 0; n < DSTATE; ++n) { ap[n] = 1.f; bc[n] = 0.f; }

    const int row0 = b * LSEQ + chunk * CLEN;
    for (int t = 0; t < CLEN; ++t) {
        const int row = row0 + t;
        const float dtv = dt[(size_t)row * DINNER + d];
        const float xv = (float)xc[(size_t)row * DINNER + d];
        const float dtx = dtv * xv;
        float Bv[DSTATE];
        load16(dbc + (size_t)row * NPADXP + DTRANK, Bv);
#pragma unroll
        for (int n = 0; n < DSTATE; ++n) {
            const float e = __expf(dtv * An[n]);
            ap[n] *= e;
            bc[n] = e * bc[n] + dtx * Bv[n];
        }
    }
    const size_t s = ((size_t)(b * DINNER + d)) * DSTATE;
#pragma unroll
    for (int n = 0; n < DSTATE; ++n) {
        Ap[(size_t)chunk * NSEQ + s + n] = ap[n];
        Bc[(size_t)chunk * NSEQ + s + n] = bc[n];
    }
}

// Mid: exclusive scan of 32 chunk summaries per (b,d,n)
__global__ void scan_mid(const float* __restrict__ Ap, const float* __restrict__ Bc,
                         float* __restrict__ Hi)
{
    int s = blockIdx.x * 256 + threadIdx.x;
    if (s >= NSEQ) return;
    float h = 0.f;
    for (int c = 0; c < NCHUNK; ++c) {
        Hi[(size_t)c * NSEQ + s] = h;
        h = Ap[(size_t)c * NSEQ + s] * h + Bc[(size_t)c * NSEQ + s];
    }
}

// Pass B: replay chunk from correct state; fuse +x*D and silu(z) gate -> ygate f16
__global__ void __launch_bounds__(32)
scan_passB(const float* __restrict__ dt, const _Float16* __restrict__ xc,
           const float* __restrict__ dbc, const float* __restrict__ A_log,
           const float* __restrict__ Dv, const _Float16* __restrict__ xz,
           const float* __restrict__ Hi, _Float16* __restrict__ yg)
{
    const int lane = threadIdx.x;
    const int chunk = blockIdx.x & (NCHUNK - 1);
    const int dblk = (blockIdx.x >> 5) % (DINNER / 32);
    const int b = blockIdx.x / (NCHUNK * (DINNER / 32));
    const int d = dblk * 32 + lane;

    float An[DSTATE];
#pragma unroll
    for (int n = 0; n < DSTATE; ++n) An[n] = -__expf(A_log[d * DSTATE + n]);
    const float Dd = Dv[d];
    const size_t s = ((size_t)(b * DINNER + d)) * DSTATE;
    float h[DSTATE];
#pragma unroll
    for (int n = 0; n < DSTATE; ++n) h[n] = Hi[(size_t)chunk * NSEQ + s + n];

    const int row0 = b * LSEQ + chunk * CLEN;
    for (int t = 0; t < CLEN; ++t) {
        const int row = row0 + t;
        const float dtv = dt[(size_t)row * DINNER + d];
        const float xv = (float)xc[(size_t)row * DINNER + d];
        const float dtx = dtv * xv;
        float Bv[DSTATE], Cv[DSTATE];
        load16(dbc + (size_t)row * NPADXP + DTRANK, Bv);
        load16(dbc + (size_t)row * NPADXP + DTRANK + DSTATE, Cv);
        float y = 0.f;
#pragma unroll
        for (int n = 0; n < DSTATE; ++n) {
            const float e = __expf(dtv * An[n]);
            h[n] = e * h[n] + dtx * Bv[n];
            y += h[n] * Cv[n];
        }
        y += xv * Dd;
        const float zv = (float)xz[(size_t)row * (2 * DINNER) + DINNER + d];
        const float sz = zv / (1.f + __expf(-zv));
        yg[(size_t)row * DINNER + d] = (_Float16)(y * sz);
    }
}

// ---------------- launcher ----------------
extern "C" void kernel_launch(void* const* d_in, const int* in_sizes, int n_in,
                              void* d_out, int out_size, void* d_ws, size_t ws_size,
                              hipStream_t stream)
{
    (void)in_sizes; (void)n_in; (void)out_size;
    const float* x         = (const float*)d_in[0];
    const float* norm_w    = (const float*)d_in[1];
    const float* norm_b    = (const float*)d_in[2];
    const float* skip      = (const float*)d_in[3];
    const float* in_proj_w = (const float*)d_in[4];
    const float* conv_w    = (const float*)d_in[5];
    const float* conv_b    = (const float*)d_in[6];
    const float* x_proj_w  = (const float*)d_in[7];
    const float* dt_proj_w = (const float*)d_in[8];
    const float* dt_proj_b = (const float*)d_in[9];
    const float* A_log     = (const float*)d_in[10];
    const float* Dvec      = (const float*)d_in[11];
    const float* out_proj_w= (const float*)d_in[12];
    const float* proj_w    = (const float*)d_in[13];
    const float* proj_b    = (const float*)d_in[14];

    // workspace carve (256B aligned)
    char* ws = (char*)d_ws;
    size_t off = 0;
    auto carve = [&](size_t bytes) { void* p = ws + off; off = (off + bytes + 255) & ~(size_t)255; return p; };
    _Float16* xn   = (_Float16*)carve((size_t)BL * DMODEL * 2);
    _Float16* w1   = (_Float16*)carve((size_t)(2*DINNER) * DMODEL * 2);
    _Float16* w2   = (_Float16*)carve((size_t)NPADXP * DINNER * 2);
    _Float16* w3   = (_Float16*)carve((size_t)DMODEL * DINNER * 2);
    _Float16* w4   = (_Float16*)carve((size_t)DOUT * DMODEL * 2);
    _Float16* xz   = (_Float16*)carve((size_t)BL * 2 * DINNER * 2);
    _Float16* xc   = (_Float16*)carve((size_t)BL * DINNER * 2);
    float*    dbc  = (float*)   carve((size_t)BL * NPADXP * 4);
    float*    dtb  = (float*)   carve((size_t)BL * DINNER * 4);
    float*    Ap   = (float*)   carve((size_t)NSEQ * NCHUNK * 4);
    float*    Bc   = (float*)   carve((size_t)NSEQ * NCHUNK * 4);
    float*    Hi   = (float*)   carve((size_t)NSEQ * NCHUNK * 4);
    _Float16* yg   = (_Float16*)carve((size_t)BL * DINNER * 2);
    // aliases for dead buffers: ym over xz (z consumed in passB), xn2 over xn (dead after G1)
    float*    ym   = (float*)xz;
    _Float16* xn2  = xn;
    (void)ws_size;

    // weights -> f16 (x_proj padded 44 -> 48 rows)
    cvt_f16<<<(2*DINNER*DMODEL + 255)/256, 256, 0, stream>>>(in_proj_w, w1, 2*DINNER*DMODEL);
    cvt_f16_pad<<<(NPADXP*DINNER + 255)/256, 256, 0, stream>>>(x_proj_w, w2, 44*DINNER, NPADXP*DINNER);
    cvt_f16<<<(DMODEL*DINNER + 255)/256, 256, 0, stream>>>(out_proj_w, w3, DMODEL*DINNER);
    cvt_f16<<<(DOUT*DMODEL + 255)/256, 256, 0, stream>>>(proj_w, w4, DOUT*DMODEL);

    // LN1 (transpose [b,c,l] -> [row,c], f16)
    ln_kernel<<<BL, 256, 0, stream>>>(x, nullptr, nullptr, norm_w, norm_b, xn);

    // G1: xz[BL,768] = xn @ in_proj_w^T      LDS: 64x192 f16 = 24KB
    gemm_wmma<4, 0><<<dim3(BL/(16*MWAVES), (2*DINNER)/64), 256,
                      64*DMODEL*2, stream>>>(
        xn, w1, xz, nullptr, BL, 2*DINNER, DMODEL, 2*DINNER);

    // conv + silu -> xc
    conv_silu<<<(BL*DINNER + 255)/256, 256, 0, stream>>>(xz, conv_w, conv_b, xc);

    // G2: dbc[BL,48] = xc @ x_proj_w^T (padded)   LDS: 48x384 f16 = 36KB
    gemm_wmma<3, 1><<<dim3(BL/(16*MWAVES), 1), 256, NPADXP*DINNER*2, stream>>>(
        xc, w2, dbc, nullptr, BL, NPADXP, DINNER, NPADXP);

    // dt = softplus(dbc[:, :12] @ dt_proj_w^T + b)
    dt_softplus<<<(BL*DINNER + 255)/256, 256, 0, stream>>>(dbc, dt_proj_w, dt_proj_b, dtb);

    // chunked selective scan
    const int scan_blocks = BATCH * (DINNER/32) * NCHUNK; // 3072
    scan_passA<<<scan_blocks, 32, 0, stream>>>(dtb, xc, dbc, A_log, Ap, Bc);
    scan_mid<<<(NSEQ + 255)/256, 256, 0, stream>>>(Ap, Bc, Hi);
    scan_passB<<<scan_blocks, 32, 0, stream>>>(dtb, xc, dbc, A_log, Dvec, xz, Hi, yg);

    // G3: ym[BL,192] = yg @ out_proj_w^T     LDS: 64x384 f16 = 48KB
    gemm_wmma<4, 1><<<dim3(BL/(16*MWAVES), DMODEL/64), 256, 64*DINNER*2, stream>>>(
        yg, w3, ym, nullptr, BL, DMODEL, DINNER, DMODEL);

    // residual + LN2 -> xn2 f16
    ln_kernel<<<BL, 256, 0, stream>>>(x, ym, skip, norm_w, norm_b, xn2);

    // G4: out[b,o,l] = xn2 @ proj_w^T + proj_b  (transposed store into d_out)
    //     LDS: 64x192 f16 = 24KB
    gemm_wmma<4, 2><<<dim3(BL/(16*MWAVES), DOUT/64), 256, 64*DMODEL*2, stream>>>(
        xn2, w4, d_out, proj_b, BL, DOUT, DMODEL, LSEQ);
}
